// BahdanauAttention_57054345560395
// MI455X (gfx1250) — matrix operand order repaired and checked by reading, not compiled, not run on hardware
//
#include <hip/hip_runtime.h>
#include <hip/hip_bf16.h>
#include <math.h>

// ---------------------------------------------------------------------------
// Bahdanau attention, MI455X (gfx1250, wave32, WMMA).
// enc_p = h_enc @ We via split-bf16 (3x v_wmma_f32_16x16x32_bf16) for ~f32
// accuracy at bf16 matrix-core rate. hi/lo conversion done ONCE at LDS staging
// (packed 2xbf16 dwords); double-buffered LDS so next tile's global loads and
// conversions overlap the current tile's WMMAs (one barrier per K-step).
//
// ws layout (floats):
//   encp   : [65536, 1024]         (256 MB)
//   decp   : [32, 1024]
//   scorep : [2, 65536]            (N-split partial scores)
//   part   : [32, 8, 1024]
// d_out: context [32,1024] then attn [32,2048] (flattened, return order).
// ---------------------------------------------------------------------------

typedef __attribute__((ext_vector_type(16))) __bf16 v16bf;
typedef __attribute__((ext_vector_type(8)))  float  v8f;

#define B_  32
#define T_  2048
#define DE_ 1024
#define H_  1024

constexpr int MT   = 64;      // enc_p rows per workgroup
constexpr int NT   = 512;     // enc_p cols per workgroup (H split in 2)
constexpr int KT   = 32;      // K per step (one bf16 WMMA depth)
constexpr int KSTEPS = DE_ / KT;
constexpr int LDA  = 17;      // A row stride in dwords (16 + 1 pad, odd)
constexpr int LDB  = 17;      // B col stride in dwords (16 + 1 pad, odd)
constexpr int THREADS = 512;  // 16 waves: 2 row-groups x 8 col-groups(64)

union F16x16 { v16bf v; unsigned u[8]; };

__device__ inline v8f wmma_bf16(v16bf a, v16bf b, v8f c) {
  // (neg_a, A, neg_b, B, c_mod, C, reuse_a, reuse_b)
  return __builtin_amdgcn_wmma_f32_16x16x32_bf16(false, a, false, b,
                                                 (short)0, c, false, false);
}

__device__ inline unsigned pack_hilo(__bf16 a, __bf16 b) {
  union { __bf16 h[2]; unsigned u; } x;
  x.h[0] = a; x.h[1] = b;
  return x.u;
}

// split f32 pair -> packed hi dword + packed lo dword
__device__ inline void split2(float f0, float f1, unsigned& hi, unsigned& lo) {
  __bf16 h0 = (__bf16)f0, h1 = (__bf16)f1;
  __bf16 l0 = (__bf16)(f0 - (float)h0);
  __bf16 l1 = (__bf16)(f1 - (float)h1);
  hi = pack_hilo(h0, h1);
  lo = pack_hilo(l0, l1);
}

// --------------------------- dec_p = h_dec@Wd + bd --------------------------
__global__ __launch_bounds__(256) void dec_proj_kernel(
    const float* __restrict__ h_dec, const float* __restrict__ Wd,
    const float* __restrict__ bd, float* __restrict__ decp) {
  int h = blockIdx.x * 256 + threadIdx.x;
  int b = blockIdx.y;
  const float* hd = h_dec + (size_t)b * DE_;
  float acc = bd[h];
  for (int k = 0; k < DE_; ++k) acc += hd[k] * Wd[(size_t)k * H_ + h];
  decp[b * H_ + h] = acc;
}

// ------------- fused: enc_p tile GEMM + bias + tanh-score reduce ------------
__global__ __launch_bounds__(THREADS) void fused_proj_score_kernel(
    const float* __restrict__ h_enc, const float* __restrict__ We,
    const float* __restrict__ be,    const float* __restrict__ Wc,
    const float* __restrict__ decp,  float* __restrict__ encp,
    float* __restrict__ scorep) {
  // packed 2xbf16 tiles (hi/lo split), double buffered
  __shared__ unsigned lAhi[2][MT * LDA];   // 2 x 4352 B
  __shared__ unsigned lAlo[2][MT * LDA];   // 2 x 4352 B
  __shared__ unsigned lBhi[2][NT * LDB];   // 2 x 34816 B
  __shared__ unsigned lBlo[2][NT * LDB];   // 2 x 34816 B
  __shared__ float    lRed[MT * 8];        // 2048 B
  // total ~158.7 KB (one WG per WGP anyway; VGPR-limited)

  const int tid  = threadIdx.x;
  const int wave = tid >> 5, lane = tid & 31;
  const int rg = wave & 1;            // row group: rows rg*32 .. +31
  const int cg = wave >> 1;           // col group: cols cg*64 .. +63 (of NT)
  const int ln = lane & 15, lh = lane >> 4;
  const int m0 = blockIdx.x * MT;     // 64-row tile never crosses batch
  const int n0 = blockIdx.y * NT;     // column half
  const int b  = m0 / T_;

  float bias[4], wc[4], dp[4];
  int   cols[4];                      // global column of this lane per ct
  #pragma unroll
  for (int ct = 0; ct < 4; ++ct) {
    int c = n0 + cg * 64 + ct * 16 + ln;
    cols[ct] = c;
    bias[ct] = be[c];
    wc[ct]   = Wc[c];
    dp[ct]   = decp[b * H_ + c];
  }

  // A-stage thread mapping: 64 rows x 16 k-pairs, 2 per thread
  const int sa_r0 = tid >> 4, sa_kp = tid & 15;     // + second at r0+32
  // B-stage thread mapping: 512 cols x 16 k-pairs, 16 per thread (c fast)
  const int sb_c = tid & (NT - 1);

  auto stage = [&](int k0, int buf) {
    // ---- A: split to bf16 hi/lo once, packed pairs ----
    #pragma unroll
    for (int it = 0; it < 2; ++it) {
      int r = sa_r0 + it * 32;
      const float* src = &h_enc[(size_t)(m0 + r) * DE_ + k0 + 2 * sa_kp];
      unsigned hi, lo;
      split2(src[0], src[1], hi, lo);
      lAhi[buf][r * LDA + sa_kp] = hi;
      lAlo[buf][r * LDA + sa_kp] = lo;
    }
    // ---- B: transposed (col-major), packed pairs ----
    #pragma unroll
    for (int kp = 0; kp < 16; ++kp) {
      float f0 = We[(size_t)(k0 + 2 * kp)     * H_ + n0 + sb_c];
      float f1 = We[(size_t)(k0 + 2 * kp + 1) * H_ + n0 + sb_c];
      unsigned hi, lo;
      split2(f0, f1, hi, lo);
      lBhi[buf][sb_c * LDB + kp] = hi;
      lBlo[buf][sb_c * LDB + kp] = lo;
    }
  };

  v8f acc[2][4] = {};

  stage(0, 0);  // prologue

  #pragma unroll 2
  for (int ks = 0; ks < KSTEPS; ++ks) {
    const int cur = ks & 1;
    __syncthreads();  // stage(cur) complete; prior compute done with cur^1
    if (ks + 1 < KSTEPS) stage((ks + 1) * KT, cur ^ 1);

    // ---- A fragments: lane half selects K {0..7,16..23} vs {8..15,24..31}
    F16x16 ahi[2], alo[2];
    const int kd = lh * 4;               // dword offset of K base
    #pragma unroll
    for (int rt = 0; rt < 2; ++rt) {
      const unsigned* ah = &lAhi[cur][(rg * 32 + rt * 16 + ln) * LDA];
      const unsigned* al = &lAlo[cur][(rg * 32 + rt * 16 + ln) * LDA];
      #pragma unroll
      for (int j = 0; j < 4; ++j) {
        ahi[rt].u[j]     = ah[kd + j];        // K kb..kb+7
        ahi[rt].u[4 + j] = ah[8 + kd + j];    // K 16+kb..16+kb+7
        alo[rt].u[j]     = al[kd + j];
        alo[rt].u[4 + j] = al[8 + kd + j];
      }
    }

    // ---- B fragments per col-tile: same K split as A ----
    const int kd2 = lh * 4;
    #pragma unroll
    for (int ct = 0; ct < 4; ++ct) {
      const int cl = cg * 64 + ct * 16 + ln;   // column within NT
      const unsigned* bh = &lBhi[cur][cl * LDB];
      const unsigned* bl = &lBlo[cur][cl * LDB];
      F16x16 bhi, blo;
      #pragma unroll
      for (int j = 0; j < 4; ++j) {
        bhi.u[j]     = bh[kd2 + j];
        bhi.u[4 + j] = bh[8 + kd2 + j];
        blo.u[j]     = bl[kd2 + j];
        blo.u[4 + j] = bl[8 + kd2 + j];
      }
      #pragma unroll
      for (int rt = 0; rt < 2; ++rt) {
        acc[rt][ct] = wmma_bf16(ahi[rt].v, bhi.v, acc[rt][ct]);  // hi*hi
        acc[rt][ct] = wmma_bf16(ahi[rt].v, blo.v, acc[rt][ct]);  // hi*lo
        acc[rt][ct] = wmma_bf16(alo[rt].v, bhi.v, acc[rt][ct]);  // lo*hi
      }
    }
  }

  // Epilogue: C/D layout -> element i, lane L: row = base+i+8*(L>>4), col = ln.
  #pragma unroll
  for (int rt = 0; rt < 2; ++rt) {
    #pragma unroll
    for (int i = 0; i < 8; ++i) {
      const int row = rg * 32 + rt * 16 + i + 8 * lh;
      float s = 0.f;
      #pragma unroll
      for (int ct = 0; ct < 4; ++ct) {
        float e = acc[rt][ct][i] + bias[ct];
        encp[(size_t)(m0 + row) * H_ + cols[ct]] = e;
        s += tanhf(dp[ct] + e) * wc[ct];
      }
      // reduce over the 16 lanes of this half (same row), keep halves apart
      s += __shfl_xor(s, 1, 32);
      s += __shfl_xor(s, 2, 32);
      s += __shfl_xor(s, 4, 32);
      s += __shfl_xor(s, 8, 32);
      if (ln == 0) lRed[row * 8 + cg] = s;
    }
  }
  __syncthreads();
  for (int r = tid; r < MT; r += THREADS) {
    float s = 0.f;
    #pragma unroll
    for (int g = 0; g < 8; ++g) s += lRed[r * 8 + g];
    scorep[(size_t)blockIdx.y * (B_ * T_) + m0 + r] = s;
  }
}

// ------------------------------- softmax over T -----------------------------
__global__ __launch_bounds__(256) void softmax_kernel(
    const float* __restrict__ scorep, float* __restrict__ attn) {
  __shared__ float red[256];
  const int b = blockIdx.x, tid = threadIdx.x;
  const float* s0 = scorep + (size_t)b * T_;
  const float* s1 = scorep + (size_t)B_ * T_ + (size_t)b * T_;
  float* a = attn + (size_t)b * T_;

  float m = -3.0e38f;
  for (int t = tid; t < T_; t += 256) m = fmaxf(m, s0[t] + s1[t]);
  red[tid] = m; __syncthreads();
  for (int o = 128; o > 0; o >>= 1) {
    if (tid < o) red[tid] = fmaxf(red[tid], red[tid + o]);
    __syncthreads();
  }
  m = red[0]; __syncthreads();

  float sum = 0.f;
  for (int t = tid; t < T_; t += 256) {
    float e = expf((s0[t] + s1[t]) - m);
    a[t] = e;
    sum += e;
  }
  red[tid] = sum; __syncthreads();
  for (int o = 128; o > 0; o >>= 1) {
    if (tid < o) red[tid] += red[tid + o];
    __syncthreads();
  }
  const float inv = 1.0f / red[0];
  for (int t = tid; t < T_; t += 256) a[t] *= inv;
}

// ---------------- context partials: sum_t attn * enc_p (chunked) ------------
__global__ __launch_bounds__(256) void ctx_partial_kernel(
    const float* __restrict__ encp, const float* __restrict__ attn,
    float* __restrict__ part) {
  const int col = blockIdx.x * 256 + threadIdx.x;  // H split into 4
  const int tc  = blockIdx.y;                      // 8 chunks of 256 t
  const int b   = blockIdx.z;
  const int t0  = tc * 256;
  const float* a  = attn + (size_t)b * T_ + t0;
  const float* ep = encp + ((size_t)b * T_ + t0) * H_ + col;
  float acc = 0.f;
  for (int t = 0; t < 256; ++t) acc += a[t] * ep[(size_t)t * H_];
  part[((size_t)(b * 8 + tc)) * H_ + col] = acc;
}

__global__ __launch_bounds__(256) void ctx_final_kernel(
    const float* __restrict__ part, float* __restrict__ ctx) {
  const int idx = blockIdx.x * 256 + threadIdx.x;  // 32768
  const int b = idx >> 10, h = idx & (H_ - 1);
  float s = 0.f;
  #pragma unroll
  for (int tc = 0; tc < 8; ++tc) s += part[((size_t)(b * 8 + tc)) * H_ + h];
  ctx[idx] = s;
}

// ---------------------------------------------------------------------------
extern "C" void kernel_launch(void* const* d_in, const int* in_sizes, int n_in,
                              void* d_out, int out_size, void* d_ws, size_t ws_size,
                              hipStream_t stream) {
  (void)in_sizes; (void)n_in; (void)out_size; (void)ws_size;
  const float* h_enc = (const float*)d_in[0];  // [32,2048,1024]
  const float* h_dec = (const float*)d_in[1];  // [32,1024]
  const float* We    = (const float*)d_in[2];  // [1024,1024]
  const float* be    = (const float*)d_in[3];  // [1024]
  const float* Wd    = (const float*)d_in[4];  // [1024,1024]
  const float* bd    = (const float*)d_in[5];  // [1024]
  const float* Wc    = (const float*)d_in[6];  // [1024,1]
  // d_in[7] = bc: additive scalar before softmax -> cancels, unused.

  float* out  = (float*)d_out;
  float* ctx  = out;                    // [32,1024]
  float* attn = out + B_ * H_;          // [32,2048]

  float* ws     = (float*)d_ws;
  float* encp   = ws;                                       // 65536*1024
  float* decp   = encp + (size_t)B_ * T_ * H_;              // 32768
  float* scorep = decp + (size_t)B_ * H_;                   // 2*65536
  float* part   = scorep + (size_t)2 * B_ * T_;             // 32*8*1024

  dec_proj_kernel<<<dim3(H_ / 256, B_), 256, 0, stream>>>(h_dec, Wd, bd, decp);
  fused_proj_score_kernel<<<dim3((B_ * T_) / MT, H_ / NT), THREADS, 0, stream>>>(
      h_enc, We, be, Wc, decp, encp, scorep);
  softmax_kernel<<<dim3(B_), 256, 0, stream>>>(scorep, attn);
  ctx_partial_kernel<<<dim3(H_ / 256, 8, B_), 256, 0, stream>>>(encp, attn, part);
  ctx_final_kernel<<<dim3((B_ * H_) / 256), 256, 0, stream>>>(part, ctx);
}
